// VariableSelectionNetwork_9010841387123
// MI455X (gfx1250) — compile-verified, matrix operand order checked by reading, not run on hardware
//
#include <hip/hip_runtime.h>

typedef __attribute__((ext_vector_type(16))) __bf16 v16bf;
typedef __attribute__((ext_vector_type(8)))  __bf16 v8bf;
typedef __attribute__((ext_vector_type(8)))  float  v8f;

static __device__ __forceinline__ unsigned short f2bf(float f) {
    unsigned int u = __float_as_uint(f);
    u += 0x7FFFu + ((u >> 16) & 1u);   // round-to-nearest-even
    return (unsigned short)(u >> 16);
}

// ---------------------------------------------------------------------------
// Prepack f_fc2_w [512,256] f32 -> bf16 in WMMA B-fragment order:
// element (nt, ks, lane, i) at flat index ((nt*8+ks)*32+lane)*16
// so each lane's fragment is one contiguous 32-byte load.
// ---------------------------------------------------------------------------
__global__ void vsn_prepack_b(const float* __restrict__ W2,
                              unsigned short* __restrict__ pb) {
    int idx = blockIdx.x * 256 + threadIdx.x;       // 0 .. 131071
    int i  = idx & 15;
    int l  = (idx >> 4) & 31;
    int ks = (idx >> 9) & 7;
    int nt = idx >> 12;                             // 0 .. 31
    int v = i >> 1, p = i & 1;
    int kLocal = (v < 4 ? 2 * v : 16 + 2 * (v - 4)) + p;
    int k = ks * 32 + ((l >> 4) << 3) + kLocal;     // 0 .. 255
    int n = nt * 16 + (l & 15);                     // 0 .. 511
    pb[idx] = f2bf(W2[n * 256 + k]);
}

// ---------------------------------------------------------------------------
// Phase 1: weight GRN + softmax. One 64-thread block per token.
// ---------------------------------------------------------------------------
__global__ __launch_bounds__(64) void vsn_weights(
    const float* __restrict__ x,
    const float* __restrict__ W1, const float* __restrict__ b1,
    const float* __restrict__ W2, const float* __restrict__ b2,
    const float* __restrict__ lng, const float* __restrict__ lnb,
    float* __restrict__ wsW) {
    __shared__ float sx[64], sa[64], sr1[64], sr2[64];
    __shared__ float s_mu, s_rstd, s_max, s_sum;
    const int n = blockIdx.x, j = threadIdx.x;

    sx[j] = x[n * 64 + j];
    __syncthreads();

    float a = b1[j];
    for (int k = 0; k < 64; ++k) a += sx[k] * W1[j * 64 + k];
    a = a > 0.f ? a : __expf(a) - 1.f;              // ELU
    sa[j] = a;
    __syncthreads();

    float g1 = b2[j], g2 = b2[j + 64];
    for (int k = 0; k < 64; ++k) {
        float ak = sa[k];
        g1 += ak * W2[j * 64 + k];
        g2 += ak * W2[(j + 64) * 64 + k];
    }
    float yv = sx[j] + g1 * (1.f / (1.f + __expf(-g2)));   // GLU + identity skip

    // LayerNorm(64)
    sr1[j] = yv; sr2[j] = yv * yv;
    __syncthreads();
    for (int off = 32; off >= 1; off >>= 1) {
        if (j < off) { sr1[j] += sr1[j + off]; sr2[j] += sr2[j + off]; }
        __syncthreads();
    }
    if (j == 0) {
        float mu = sr1[0] * (1.f / 64.f);
        float var = sr2[0] * (1.f / 64.f) - mu * mu;
        s_mu = mu; s_rstd = rsqrtf(var + 1e-5f);
    }
    __syncthreads();
    float z = (yv - s_mu) * s_rstd * lng[j] + lnb[j];

    // softmax(64)
    sr1[j] = z;
    __syncthreads();
    for (int off = 32; off >= 1; off >>= 1) {
        if (j < off) sr1[j] = fmaxf(sr1[j], sr1[j + off]);
        __syncthreads();
    }
    if (j == 0) s_max = sr1[0];
    __syncthreads();
    float e = __expf(z - s_max);
    sr1[j] = e;
    __syncthreads();
    for (int off = 32; off >= 1; off >>= 1) {
        if (j < off) sr1[j] += sr1[j + off];
        __syncthreads();
    }
    if (j == 0) s_sum = sr1[0];
    __syncthreads();
    wsW[n * 64 + j] = e / s_sum;
}

// ---------------------------------------------------------------------------
// Phase 2: fused feature GRN (WMMA GEMM) + GLU + LN + weighted feature sum.
// One workgroup per token: 128 threads = 4 waves, M = 64 rows (all features),
// K = 256, N = 512. Wave wv owns g-columns [wv*64, wv*64+64) and
// [256+wv*64, ...) (paired for in-register GLU), x 4 row-tiles.
// Each B fragment is loaded once and reused by 4 WMMAs (4x less L2 traffic).
// The token's full feature-weighted output is produced in-block: no atomics.
// ---------------------------------------------------------------------------
#define HP 264   // padded row stride (ushort) for h tile; multiple of 8
__global__ __launch_bounds__(128) void vsn_main(
    const float* __restrict__ x,
    const float* __restrict__ w1, const float* __restrict__ b1,
    const unsigned short* __restrict__ pb,
    const float* __restrict__ b2,
    const float* __restrict__ lng, const float* __restrict__ lnb,
    const float* __restrict__ skw, const float* __restrict__ skb,
    const float* __restrict__ wsW,
    float* __restrict__ out) {
    __shared__ __align__(16) unsigned short lds_h[64 * HP];
    __shared__ float lds_x[64], lds_wt[64], lds_mu[64], lds_rstd[64];
    __shared__ float lds_rs[4][64], lds_rq[4][64];

    const int tid  = threadIdx.x;
    const int nIdx = blockIdx.x;       // token
    const int row0 = nIdx * 64;        // global row = n*64 + f

    if (tid < 64) {
        lds_x[tid]  = x[row0 + tid];
        lds_wt[tid] = wsW[row0 + tid];
    }
    __syncthreads();

    // Stage h[64][256] = ELU(x_m * w1 + b1) as bf16 (A operand source)
    for (int idx = tid; idx < 64 * 256; idx += 128) {
        int m = idx >> 8, k = idx & 255;
        float v = lds_x[m] * w1[k] + b1[k];
        v = v > 0.f ? v : __expf(v) - 1.f;
        lds_h[m * HP + k] = f2bf(v);
    }
    __syncthreads();

    const int wv    = tid >> 5;
    const int lane  = tid & 31;
    const int lhalf = lane >> 4;       // 0/1
    const int ln16  = lane & 15;

    v8f acc[8][4];                     // [col tile][row tile]
    #pragma unroll
    for (int t = 0; t < 8; ++t)
        #pragma unroll
        for (int rt = 0; rt < 4; ++rt)
            #pragma unroll
            for (int r = 0; r < 8; ++r) acc[t][rt][r] = 0.f;

    // K loop: 8 steps of K=32 bf16 WMMA; 8 col-tiles x 4 row-tiles per step
    #pragma unroll
    for (int ks = 0; ks < 8; ++ks) {
        int base = ks * 32 + (lhalf << 3);
        v16bf A[4];
        #pragma unroll
        for (int rt = 0; rt < 4; ++rt) {
            int mrow = rt * 16 + ln16;
            v8bf alo = *(const v8bf*)&lds_h[mrow * HP + base];
            v8bf ahi = *(const v8bf*)&lds_h[mrow * HP + base + 16];
            #pragma unroll
            for (int i = 0; i < 8; ++i) { A[rt][i] = alo[i]; A[rt][i + 8] = ahi[i]; }
        }
        #pragma unroll
        for (int t = 0; t < 8; ++t) {
            int nt = (t < 4) ? (wv * 4 + t) : (16 + wv * 4 + (t - 4));
            const v16bf B = *(const v16bf*)&pb[(((nt * 8 + ks) * 32) + lane) * 16];
            #pragma unroll
            for (int rt = 0; rt < 4; ++rt) {
                acc[t][rt] = __builtin_amdgcn_wmma_f32_16x16x32_bf16(
                    false, A[rt], false, B, (short)0, acc[t][rt], false, false);
            }
        }
    }

    // + fc2 bias
    #pragma unroll
    for (int t = 0; t < 8; ++t) {
        int coln = ((t < 4) ? (wv * 64 + 16 * t) : (256 + wv * 64 + 16 * (t - 4))) + ln16;
        float bias = b2[coln];
        #pragma unroll
        for (int rt = 0; rt < 4; ++rt)
            #pragma unroll
            for (int r = 0; r < 8; ++r) acc[t][rt][r] += bias;
    }

    // GLU + linear skip: y = (x*skw+skb) + g1*sigmoid(g2); tiles t and t+4 pair
    #pragma unroll
    for (int t = 0; t < 4; ++t) {
        int jc = wv * 64 + 16 * t + ln16;           // 0..255
        float sw = skw[jc], sb = skb[jc];
        #pragma unroll
        for (int rt = 0; rt < 4; ++rt) {
            #pragma unroll
            for (int r = 0; r < 8; ++r) {
                int m = rt * 16 + r + (lhalf << 3);
                float g1 = acc[t][rt][r], g2 = acc[t + 4][rt][r];
                float sg = 1.f / (1.f + __expf(-g2));
                acc[t][rt][r] = lds_x[m] * sw + sb + g1 * sg;
            }
        }
    }

    // LayerNorm(256) row statistics: in-wave 16-lane reduction + LDS combine
    #pragma unroll
    for (int rt = 0; rt < 4; ++rt) {
        float ps[8], pq[8];
        #pragma unroll
        for (int r = 0; r < 8; ++r) {
            float s = 0.f, q = 0.f;
            #pragma unroll
            for (int t = 0; t < 4; ++t) { float yv = acc[t][rt][r]; s += yv; q += yv * yv; }
            ps[r] = s; pq[r] = q;
        }
        #pragma unroll
        for (int off = 1; off < 16; off <<= 1) {
            #pragma unroll
            for (int r = 0; r < 8; ++r) {
                ps[r] += __shfl_xor(ps[r], off, 32);
                pq[r] += __shfl_xor(pq[r], off, 32);
            }
        }
        if (ln16 == 0) {
            #pragma unroll
            for (int r = 0; r < 8; ++r) {
                int m = rt * 16 + r + (lhalf << 3);
                lds_rs[wv][m] = ps[r];
                lds_rq[wv][m] = pq[r];
            }
        }
    }
    __syncthreads();
    if (tid < 64) {
        float S = 0.f, Q = 0.f;
        for (int w = 0; w < 4; ++w) { S += lds_rs[w][tid]; Q += lds_rq[w][tid]; }
        float mu  = S * (1.f / 256.f);
        float var = Q * (1.f / 256.f) - mu * mu;
        lds_mu[tid]   = mu;
        lds_rstd[tid] = rsqrtf(var + 1e-5f);
    }
    __syncthreads();

    // proc = LN(y); out[n, jc] = sum over all 64 rows of wt[m]*proc[m, jc]
    #pragma unroll
    for (int t = 0; t < 4; ++t) {
        int jc = wv * 64 + 16 * t + ln16;
        float lg = lng[jc], lb = lnb[jc];
        float contrib = 0.f;
        #pragma unroll
        for (int rt = 0; rt < 4; ++rt) {
            #pragma unroll
            for (int r = 0; r < 8; ++r) {
                int m = rt * 16 + r + (lhalf << 3);
                float proc = (acc[t][rt][r] - lds_mu[m]) * lds_rstd[m] * lg + lb;
                contrib += lds_wt[m] * proc;
            }
        }
        contrib += __shfl_xor(contrib, 16, 32);     // fold other 32 rows' half
        if (lane < 16) out[nIdx * 256 + jc] = contrib;
    }
}

// ---------------------------------------------------------------------------
extern "C" void kernel_launch(void* const* d_in, const int* in_sizes, int n_in,
                              void* d_out, int out_size, void* d_ws, size_t ws_size,
                              hipStream_t stream) {
    const float* x       = (const float*)d_in[0];   // [B,S,64]
    const float* w_fc1_w = (const float*)d_in[1];   // [64,64]
    const float* w_fc1_b = (const float*)d_in[2];   // [64]
    const float* w_fc2_w = (const float*)d_in[3];   // [128,64]
    const float* w_fc2_b = (const float*)d_in[4];   // [128]
    const float* w_ln_g  = (const float*)d_in[5];   // [64]
    const float* w_ln_b  = (const float*)d_in[6];   // [64]
    const float* f_fc1_w = (const float*)d_in[7];   // [256,1]
    const float* f_fc1_b = (const float*)d_in[8];   // [256]
    const float* f_fc2_w = (const float*)d_in[9];   // [512,256]
    const float* f_fc2_b = (const float*)d_in[10];  // [512]
    const float* f_ln_g  = (const float*)d_in[11];  // [256]
    const float* f_ln_b  = (const float*)d_in[12];  // [256]
    const float* f_skip_w= (const float*)d_in[13];  // [256,1]
    const float* f_skip_b= (const float*)d_in[14];  // [256]
    float* out = (float*)d_out;

    const int NF = in_sizes[0];        // B*S*F = 262144
    const int N  = NF / 64;            // tokens = 4096

    float* wsW = (float*)d_ws;                                       // N*64 f32
    unsigned short* pb =
        (unsigned short*)((char*)d_ws + (size_t)NF * sizeof(float)); // 512*256 bf16

    vsn_prepack_b<<<512, 256, 0, stream>>>(f_fc2_w, pb);
    vsn_weights<<<N, 64, 0, stream>>>(x, w_fc1_w, w_fc1_b, w_fc2_w, w_fc2_b,
                                      w_ln_g, w_ln_b, wsW);
    vsn_main<<<N, 128, 0, stream>>>(x, f_fc1_w, f_fc1_b, pb, f_fc2_b,
                                    f_ln_g, f_ln_b, f_skip_w, f_skip_b,
                                    wsW, out);
    (void)n_in; (void)ws_size; (void)out_size;
}
